// kNN_57878979280927
// MI455X (gfx1250) — matrix operand order repaired and checked by reading, not compiled, not run on hardware
//
#include <hip/hip_runtime.h>
#include <math.h>

// ---------------- configuration ----------------
constexpr int NQ    = 4096;   // queries
constexpr int NT    = 65536;  // train points
constexpr int DIM   = 512;    // feature dim
constexpr int KTOP  = 16;     // top-k
constexpr int NPART = 16;     // train partitions (parallelism on topk)
constexpr int PARTN = NT / NPART;  // 4096 train cols per partition
constexpr int QT    = 32;     // query tile rows per block (2 WMMA m-tiles)
constexpr int MT    = QT / 16;     // m-tiles per wave
constexpr int CT    = 64;     // train cols per block iteration (4 waves x 16)
constexpr int NKS   = DIM / 32;    // 16 k-steps of 32

typedef __bf16 v16bf __attribute__((ext_vector_type(16)));
typedef float  v8f   __attribute__((ext_vector_type(8)));

union FragBF { unsigned u[8]; uint4 q[2]; v16bf v; };

__device__ __forceinline__ unsigned short bfbits(float f) {
  return __builtin_bit_cast(unsigned short, (__bf16)f);   // hw v_cvt, RNE
}

// keep a sorted-descending top-16 in registers
__device__ __forceinline__ void topk_insert(float (&tv)[KTOP], int (&ti)[KTOP],
                                            float v, int idx) {
  if (v > tv[KTOP - 1]) {
    tv[KTOP - 1] = v; ti[KTOP - 1] = idx;
#pragma unroll
    for (int e = KTOP - 1; e > 0; --e) {
      if (tv[e] > tv[e - 1]) {
        float t = tv[e]; tv[e] = tv[e - 1]; tv[e - 1] = t;
        int   u = ti[e]; ti[e] = ti[e - 1]; ti[e - 1] = u;
      }
    }
  }
}

// ---- kernel 0: convert x_train -> bf16 hi/lo planes + all squared norms ----
__global__ __launch_bounds__(256) void knn_prep(const float* __restrict__ xte,
                                                const float* __restrict__ xtr,
                                                float* __restrict__ sqTest,
                                                float* __restrict__ sqTrain,
                                                unsigned short* __restrict__ trHi,
                                                unsigned short* __restrict__ trLo) {
  int gw   = (blockIdx.x * blockDim.x + threadIdx.x) >> 5;  // global wave id
  int lane = threadIdx.x & 31;
  if (gw >= NT + NQ) return;
  const bool isTrain = gw < NT;
  const float* p = isTrain ? (xtr + (size_t)gw * DIM)
                           : (xte + (size_t)(gw - NT) * DIM);
  const float4* p4 = (const float4*)p;
  float s = 0.0f;
#pragma unroll
  for (int i = 0; i < DIM / 128; ++i) {          // 4 x float4 per lane
    float4 v = p4[i * 32 + lane];
    s += v.x * v.x + v.y * v.y + v.z * v.z + v.w * v.w;
    if (isTrain) {
      int k = (i * 32 + lane) * 4;
      __bf16 h0 = (__bf16)v.x, h1 = (__bf16)v.y, h2 = (__bf16)v.z, h3 = (__bf16)v.w;
      float r0 = v.x - (float)h0, r1 = v.y - (float)h1;
      float r2 = v.z - (float)h2, r3 = v.w - (float)h3;
      uint2 uh, ul;
      uh.x = (unsigned)__builtin_bit_cast(unsigned short, h0) |
             ((unsigned)__builtin_bit_cast(unsigned short, h1) << 16);
      uh.y = (unsigned)__builtin_bit_cast(unsigned short, h2) |
             ((unsigned)__builtin_bit_cast(unsigned short, h3) << 16);
      ul.x = (unsigned)bfbits(r0) | ((unsigned)bfbits(r1) << 16);
      ul.y = (unsigned)bfbits(r2) | ((unsigned)bfbits(r3) << 16);
      *(uint2*)(trHi + (size_t)gw * DIM + k) = uh;
      *(uint2*)(trLo + (size_t)gw * DIM + k) = ul;
    }
  }
#pragma unroll
  for (int off = 16; off > 0; off >>= 1) s += __shfl_xor(s, off, 32);
  if (lane == 0) {
    if (isTrain) sqTrain[gw] = s; else sqTest[gw - NT] = s;
  }
}

// ---- kernel 1: fused WMMA distances + partial top-k ----
__global__ __launch_bounds__(128) void knn_main(const unsigned short* __restrict__ trHi,
                                                const unsigned short* __restrict__ trLo,
                                                const float* __restrict__ xte,
                                                const float* __restrict__ sqTest,
                                                const float* __restrict__ sqTrain,
                                                float* __restrict__ partV,
                                                int*   __restrict__ partI) {
  // A tiles pre-shuffled into WMMA fragment order: [m][kstep][half][8 dwords]
  __shared__ __align__(16) unsigned sAhi[QT][NKS][2][8];   // 32 KB
  __shared__ __align__(16) unsigned sAlo[QT][NKS][2][8];   // 32 KB
  __shared__ float sQ[QT];
  __shared__ float sDist[QT][CT + 4];                      // padded vs bank conflicts
  __shared__ float sPV[128][KTOP];
  __shared__ int   sPI[128][KTOP];

  const int tid  = threadIdx.x;
  const int wave = tid >> 5;
  const int lane = tid & 31;
  const int hv   = lane >> 4;     // half-wave select
  const int lm   = lane & 15;
  const int q0   = blockIdx.x * QT;
  const int part = blockIdx.y;
  const int pbase = part * PARTN;

  // stage A tile: f32 -> hi/lo bf16, shuffled into fragment layout
  // one u32 (pair of consecutive k) per iteration
  const float2* xt2 = (const float2*)xte;
  for (int i = tid; i < QT * (DIM / 2); i += 128) {
    int m  = i >> 8;            // / (DIM/2)
    int pr = i & 255;           // pair index within row
    int kk = pr >> 4;           // k-step
    int p  = pr & 15;           // pair within 32-K block
    int hvv, vv;
    if (p < 4)       { hvv = 0; vv = p; }
    else if (p < 8)  { hvv = 1; vv = p - 4; }
    else if (p < 12) { hvv = 0; vv = p - 4; }
    else             { hvv = 1; vv = p - 8; }
    float2 f = xt2[(size_t)(q0 + m) * (DIM / 2) + pr];
    __bf16 h0 = (__bf16)f.x, h1 = (__bf16)f.y;
    float  r0 = f.x - (float)h0, r1 = f.y - (float)h1;
    sAhi[m][kk][hvv][vv] = (unsigned)__builtin_bit_cast(unsigned short, h0) |
                           ((unsigned)__builtin_bit_cast(unsigned short, h1) << 16);
    sAlo[m][kk][hvv][vv] = (unsigned)bfbits(r0) | ((unsigned)bfbits(r1) << 16);
  }
  if (tid < QT) sQ[tid] = sqTest[q0 + tid];

  float tv[KTOP]; int ti[KTOP];
#pragma unroll
  for (int e = 0; e < KTOP; ++e) { tv[e] = -1.0f; ti[e] = -1; }

  __syncthreads();

  const int row  = tid & 31;   // topk-scan mapping: 4 slots x 32 rows
  const int slot = tid >> 5;

  for (int c0 = 0; c0 < PARTN; c0 += CT) {
    const int j = pbase + c0 + wave * 16 + lm;   // train column this lane owns
    if (c0 + CT < PARTN) {
      __builtin_prefetch(trHi + (size_t)(j + CT) * DIM, 0, 0);  // global_prefetch_b8
      __builtin_prefetch(trLo + (size_t)(j + CT) * DIM, 0, 0);
    }
    const unsigned short* bh = trHi + (size_t)j * DIM;
    const unsigned short* bl = trLo + (size_t)j * DIM;
    const float sqT = sqTrain[j];

    v8f acc[MT];
#pragma unroll
    for (int mt = 0; mt < MT; ++mt) acc[mt] = (v8f){};

#pragma unroll 2
    for (int kk = 0; kk < NKS; ++kk) {
      const int kb = kk * 32 + 8 * hv;
      FragBF bHi, bLo;
      bHi.q[0] = *(const uint4*)(bh + kb);
      bHi.q[1] = *(const uint4*)(bh + kb + 16);
      bLo.q[0] = *(const uint4*)(bl + kb);
      bLo.q[1] = *(const uint4*)(bl + kb + 16);
#pragma unroll
      for (int mt = 0; mt < MT; ++mt) {
        FragBF aHi, aLo;
        aHi.q[0] = *(const uint4*)&sAhi[mt * 16 + lm][kk][hv][0];
        aHi.q[1] = *(const uint4*)&sAhi[mt * 16 + lm][kk][hv][4];
        aLo.q[0] = *(const uint4*)&sAlo[mt * 16 + lm][kk][hv][0];
        aLo.q[1] = *(const uint4*)&sAlo[mt * 16 + lm][kk][hv][4];
        acc[mt] = __builtin_amdgcn_wmma_f32_16x16x32_bf16(false, aHi.v, false, bHi.v,
                                                          (short)0, acc[mt], false, false);
        acc[mt] = __builtin_amdgcn_wmma_f32_16x16x32_bf16(false, aHi.v, false, bLo.v,
                                                          (short)0, acc[mt], false, false);
        acc[mt] = __builtin_amdgcn_wmma_f32_16x16x32_bf16(false, aLo.v, false, bHi.v,
                                                          (short)0, acc[mt], false, false);
      }
    }

    // ---- dot -> distance, into LDS tile ----
#pragma unroll
    for (int mt = 0; mt < MT; ++mt) {
#pragma unroll
      for (int r8 = 0; r8 < 8; ++r8) {
        int m = mt * 16 + r8 + 8 * hv;          // C layout: VGPR r -> M=r / r+8
        float d2 = sQ[m] + sqT - 2.0f * acc[mt][r8];
        sDist[m][wave * 16 + lm] = sqrtf(fmaxf(d2, 0.0f));
      }
    }
    __syncthreads();

    // ---- running top-k: 4 threads per query row scan 16 cols each ----
#pragma unroll
    for (int cc = 0; cc < 16; ++cc) {
      int col = slot * 16 + cc;
      topk_insert(tv, ti, sDist[row][col], pbase + c0 + col);
    }
    __syncthreads();
  }

  // ---- merge 4 private lists per row -> partition top-16 ----
#pragma unroll
  for (int e = 0; e < KTOP; ++e) { sPV[tid][e] = tv[e]; sPI[tid][e] = ti[e]; }
  __syncthreads();
  if (tid < QT) {
    float mv[KTOP]; int mi[KTOP];
#pragma unroll
    for (int e = 0; e < KTOP; ++e) { mv[e] = -1.0f; mi[e] = -1; }
    for (int s = 0; s < 4; ++s)
#pragma unroll
      for (int e = 0; e < KTOP; ++e)
        topk_insert(mv, mi, sPV[s * 32 + tid][e], sPI[s * 32 + tid][e]);
    size_t base = (((size_t)(q0 + tid)) * NPART + part) * KTOP;
#pragma unroll
    for (int e = 0; e < KTOP; ++e) { partV[base + e] = mv[e]; partI[base + e] = mi[e]; }
  }
}

// ---- kernel 2: merge partitions -> final top-16 ----
__global__ __launch_bounds__(256) void knn_merge(const float* __restrict__ partV,
                                                 const int*   __restrict__ partI,
                                                 float* __restrict__ out) {
  int q = blockIdx.x * blockDim.x + threadIdx.x;
  if (q >= NQ) return;
  float tv[KTOP]; int ti[KTOP];
#pragma unroll
  for (int e = 0; e < KTOP; ++e) { tv[e] = -1.0f; ti[e] = -1; }
  const float* pv = partV + (size_t)q * NPART * KTOP;
  const int*   pi = partI + (size_t)q * NPART * KTOP;
  for (int s = 0; s < NPART * KTOP; ++s) topk_insert(tv, ti, pv[s], pi[s]);
#pragma unroll
  for (int e = 0; e < KTOP; ++e) {
    out[(size_t)q * KTOP + e] = tv[e];                            // values
    out[(size_t)NQ * KTOP + (size_t)q * KTOP + e] = (float)ti[e]; // indices
  }
}

// ---------------- launcher ----------------
extern "C" void kernel_launch(void* const* d_in, const int* in_sizes, int n_in,
                              void* d_out, int out_size, void* d_ws, size_t ws_size,
                              hipStream_t stream) {
  (void)in_sizes; (void)n_in; (void)out_size; (void)ws_size;
  const float* x_test  = (const float*)d_in[0];
  const float* x_train = (const float*)d_in[1];
  // d_in[2] is k (==16), hardcoded as KTOP.

  float* sqTest  = (float*)d_ws;                 // 4096 f32
  float* sqTrain = sqTest + NQ;                  // 65536 f32
  float* partV   = sqTrain + NT;                 // NQ*NPART*KTOP f32 (4 MB)
  int*   partI   = (int*)(partV + (size_t)NQ * NPART * KTOP);          // 4 MB
  unsigned short* trHi = (unsigned short*)(partI + (size_t)NQ * NPART * KTOP); // 64 MB
  unsigned short* trLo = trHi + (size_t)NT * DIM;                      // 64 MB

  knn_prep<<<dim3((NT + NQ) / 8), 256, 0, stream>>>(x_test, x_train, sqTest, sqTrain,
                                                    trHi, trLo);
  knn_main<<<dim3(NQ / QT, NPART), 128, 0, stream>>>(trHi, trLo, x_test, sqTest, sqTrain,
                                                     partV, partI);
  knn_merge<<<dim3(NQ / 256), 256, 0, stream>>>(partV, partI, (float*)d_out);
}
